// feat_self_30726196036057
// MI455X (gfx1250) — matrix operand order repaired and checked by reading, not compiled, not run on hardware
//
#include <hip/hip_runtime.h>
#include <hip/hip_bf16.h>

// ---------------------------------------------------------------------------
// Problem sizes (fixed by the reference)
// ---------------------------------------------------------------------------
#define BB  64
#define CC  2048
#define CI  1024
#define NN  512          // H*W = 32*16
#define BN_EPS 1e-5f

// ---------------------------------------------------------------------------
// CDNA5 WMMA types
// ---------------------------------------------------------------------------
typedef __attribute__((ext_vector_type(16))) __bf16 v16bf;
typedef __attribute__((ext_vector_type(8)))  float  v8f;

union FragBF {
    v16bf v;
    unsigned short s[16];
};
union Stage16 {            // 16 bf16 staged in registers
    uint4 u[2];
    unsigned short s[16];
};

// f32 -> bf16 round-to-nearest-even
static __device__ __forceinline__ unsigned short f2bf(float f) {
    unsigned u = __float_as_uint(f);
    u += 0x7FFFu + ((u >> 16) & 1u);
    return (unsigned short)(u >> 16);
}
static __device__ __forceinline__ float bf2f(unsigned short s) {
    return __uint_as_float(((unsigned)s) << 16);
}

// ---------------------------------------------------------------------------
// CDNA5 async global -> LDS copy (GLOBAL_LOAD_ASYNC_TO_LDS_B128, ASYNCcnt)
// Builtin signature (from hipcc diagnostic): param0 is a non-const global
// pointer to `int __attribute__((vector_size(16)))`; 4 args total.
// ---------------------------------------------------------------------------
#if defined(__has_builtin)
#if __has_builtin(__builtin_amdgcn_global_load_async_to_lds_b128)
#define ASYNC_VIA_BUILTIN 1
#endif
#endif

typedef int v4i_vs __attribute__((vector_size(16)));
typedef __attribute__((address_space(1))) v4i_vs* gv4i_p;
typedef __attribute__((address_space(3))) v4i_vs* lv4i_p;

static __device__ __forceinline__ void async_copy_b128(const void* g, void* l) {
#ifdef ASYNC_VIA_BUILTIN
    __builtin_amdgcn_global_load_async_to_lds_b128(
        (gv4i_p)g, (lv4i_p)l, 0, 0);
#else
    unsigned lds = (unsigned)(unsigned long long)
        (__attribute__((address_space(3))) void*)l;
    unsigned long long ga = (unsigned long long)g;
    asm volatile("global_load_async_to_lds_b128 %0, %1, off"
                 :: "v"(lds), "v"(ga) : "memory");
#endif
}

static __device__ __forceinline__ void wait_async0() {
#if defined(__has_builtin)
#if __has_builtin(__builtin_amdgcn_s_wait_asynccnt)
    __builtin_amdgcn_s_wait_asynccnt(0);
#else
    asm volatile("s_wait_asynccnt 0x0" ::: "memory");
#endif
#else
    asm volatile("s_wait_asynccnt 0x0" ::: "memory");
#endif
}

// ---------------------------------------------------------------------------
// f32 -> bf16 conversion kernel (4 elements / thread; sizes are multiples of 4)
// ---------------------------------------------------------------------------
__global__ __launch_bounds__(256) void cvt_f32_bf16_kernel(
    const float* __restrict__ in, unsigned short* __restrict__ out, long long n)
{
    long long i = ((long long)blockIdx.x * 256 + threadIdx.x) * 4;
    if (i + 3 < n) {
        float4 v = *(const float4*)(in + i);
        ushort4 r;
        r.x = f2bf(v.x); r.y = f2bf(v.y); r.z = f2bf(v.z); r.w = f2bf(v.w);
        *(ushort4*)(out + i) = r;
    }
}

// ---------------------------------------------------------------------------
// Generic tiled WMMA GEMM:  D[M, Ncols] = A * B (+ bias[m]),  f32 accumulate.
//   OPA == 0 : A[m][k] = Ag[m*lda + k]        (row-major MxK)   -> async to LDS
//   OPA == 1 : A[m][k] = Ag[k*lda + m]        (transposed read) -> reg staging
//   OPB == 0 : B[k][n] = Bg[k*ldb + n]        (row-major KxN)   -> reg staging
//   OPB == 1 : B[k][n] = Bg[n*ldb + k]        (transposed read) -> async to LDS
// Block tile 128(M) x 128(N), K-step 32, double-buffered LDS.
// 256 threads = 8 waves (wave32) arranged 4(M) x 2(N); wave tile 32x64 =
// 2x4 WMMA 16x16 tiles -> 8 x v_wmma_f32_16x16x32_bf16 per wave per K-step.
// Requires: M % 128 == 0, Ncols % 128 == 0, K % 32 == 0 (true for all uses).
// ---------------------------------------------------------------------------
#define TMB 128
#define TNB 128
#define TK  32
#define LDSS 40   // LDS row stride in bf16 elements (80 B: 16B-aligned, padded)
#define ATILE (TMB * LDSS)
#define BTILE (TNB * LDSS)

template <int OPA, int OPB, int OUT_BF16, int BIAS>
__global__ __launch_bounds__(256, 1) void gemm_wmma_kernel(
    const unsigned short* __restrict__ Ag, long long strideA, int lda,
    const unsigned short* __restrict__ Bg, long long strideB, int ldb,
    void* __restrict__ Dg, long long strideD, int ldd,
    const float* __restrict__ bias,
    int M, int Ncols, int K)
{
    __shared__ unsigned short As[2 * ATILE];
    __shared__ unsigned short Bs[2 * BTILE];

    const int tid  = threadIdx.x;
    const int lane = tid & 31;
    const int wave = tid >> 5;
    const int half = lane >> 4;     // 0: lanes 0-15, 1: lanes 16-31
    const int l16  = lane & 15;

    const int batch = blockIdx.z;
    const unsigned short* A = Ag + (size_t)batch * strideA;
    const unsigned short* B = Bg + (size_t)batch * strideB;

    const int m0 = blockIdx.y * TMB;
    const int n0 = blockIdx.x * TNB;

    const int wm = (wave >> 1) * 32;   // wave M offset: 0,32,64,96
    const int wn = (wave & 1) * 64;    // wave N offset: 0,64

    constexpr bool ASYNC_A = (OPA == 0);   // k-contiguous -> LDS-direct
    constexpr bool ASYNC_B = (OPB == 1);   // k-contiguous -> LDS-direct

    // per-thread staging coordinates (16 bf16 per thread per tile)
    const int aR  = (OPA == 0) ? (tid >> 1) : (tid >> 3);        // row / k
    const int aC  = (OPA == 0) ? ((tid & 1) * 16) : ((tid & 7) * 16);
    const int bR  = (OPB == 0) ? (tid >> 3) : (tid >> 1);        // k / n
    const int bC  = (OPB == 0) ? ((tid & 7) * 16) : ((tid & 1) * 16);

    auto srcA = [&](int k0) -> const unsigned short* {
        return (OPA == 0) ? A + (size_t)(m0 + aR) * lda + k0 + aC
                          : A + (size_t)(k0 + aR) * lda + m0 + aC;
    };
    auto srcB = [&](int k0) -> const unsigned short* {
        return (OPB == 0) ? B + (size_t)(k0 + bR) * ldb + n0 + bC
                          : B + (size_t)(n0 + bR) * ldb + k0 + bC;
    };

    // async (LDS-direct) staging: two B128 per thread per tile
    auto asyncA = [&](int k0, unsigned short* buf) {
        const unsigned short* s = srcA(k0);
        unsigned short* d = &buf[aR * LDSS + aC];
        async_copy_b128(s, d);
        async_copy_b128(s + 8, d + 8);
    };
    auto asyncB = [&](int k0, unsigned short* buf) {
        const unsigned short* s = srcB(k0);
        unsigned short* d = &buf[bR * LDSS + bC];
        async_copy_b128(s, d);
        async_copy_b128(s + 8, d + 8);
    };

    // register staging (transpose-scatter paths)
    auto loadA = [&](int k0, Stage16& r) {
        const unsigned short* s = srcA(k0);
        r.u[0] = *(const uint4*)(s);
        r.u[1] = *(const uint4*)(s + 8);
    };
    auto loadB = [&](int k0, Stage16& r) {
        const unsigned short* s = srcB(k0);
        r.u[0] = *(const uint4*)(s);
        r.u[1] = *(const uint4*)(s + 8);
    };
    auto storeA = [&](const Stage16& r, unsigned short* buf) {   // scatter m rows
#pragma unroll
        for (int i = 0; i < 16; ++i) buf[(aC + i) * LDSS + aR] = r.s[i];
    };
    auto storeB = [&](const Stage16& r, unsigned short* buf) {   // scatter n rows
#pragma unroll
        for (int i = 0; i < 16; ++i) buf[(bC + i) * LDSS + bR] = r.s[i];
    };

    v8f acc[2][4];
    const v8f vzero = {0.f, 0.f, 0.f, 0.f, 0.f, 0.f, 0.f, 0.f};
#pragma unroll
    for (int s = 0; s < 2; ++s)
#pragma unroll
        for (int t = 0; t < 4; ++t) acc[s][t] = vzero;

    // ---- prologue: stage K-tile 0 into buffer 0 ----
    if constexpr (ASYNC_A) { asyncA(0, As); }
    else { Stage16 r; loadA(0, r); storeA(r, As); }
    if constexpr (ASYNC_B) { asyncB(0, Bs); }
    else { Stage16 r; loadB(0, r); storeB(r, Bs); }

    int cur = 0;
    for (int k0 = 0; k0 < K; k0 += TK) {
        if constexpr (ASYNC_A || ASYNC_B) wait_async0();  // our LDS writes done
        __syncthreads();   // staged tile `cur` visible to all waves

        const bool hasNext = (k0 + TK) < K;
        Stage16 ra, rb;
        if (hasNext) {
            // async copies stream into the idle buffer while we do WMMAs;
            // register-staged matrices issue their global loads here and
            // store to LDS after the WMMAs.
            if constexpr (ASYNC_A) asyncA(k0 + TK, As + (cur ^ 1) * ATILE);
            else                   loadA(k0 + TK, ra);
            if constexpr (ASYNC_B) asyncB(k0 + TK, Bs + (cur ^ 1) * BTILE);
            else                   loadB(k0 + TK, rb);
        }
        if (k0 + 2 * TK < K) {   // prefetch two tiles ahead into GL2
            __builtin_prefetch(srcA(k0 + 2 * TK), 0, 3);
            __builtin_prefetch(srcB(k0 + 2 * TK), 0, 3);
        }

        // ---- build fragments (ISA 7.12.2 bf16 16x32 A / 32x16 B layouts) ----
        const unsigned short* Ab = As + cur * ATILE;
        const unsigned short* Bb = Bs + cur * BTILE;
        FragBF fa[2], fb[4];
#pragma unroll
        for (int s = 0; s < 2; ++s) {
            const unsigned short* ap = &Ab[(wm + s * 16 + l16) * LDSS];
            *(uint4*)&fa[s].s[0] = *(const uint4*)(ap + half * 8);
            *(uint4*)&fa[s].s[8] = *(const uint4*)(ap + 16 + half * 8);
        }
#pragma unroll
        for (int t = 0; t < 4; ++t) {
            const unsigned short* bp = &Bb[(wn + t * 16 + l16) * LDSS + half * 8];
            *(uint4*)&fb[t].s[0] = *(const uint4*)(bp);
            *(uint4*)&fb[t].s[8] = *(const uint4*)(bp + 16);
        }

        // ---- 8 x v_wmma_f32_16x16x32_bf16 ----
#pragma unroll
        for (int s = 0; s < 2; ++s)
#pragma unroll
            for (int t = 0; t < 4; ++t)
                acc[s][t] = __builtin_amdgcn_wmma_f32_16x16x32_bf16(
                    false, fa[s].v, false, fb[t].v,
                    (short)0, acc[s][t], false, false);

        if (hasNext) {
            if constexpr (!ASYNC_A) storeA(ra, As + (cur ^ 1) * ATILE);
            if constexpr (!ASYNC_B) storeB(rb, Bs + (cur ^ 1) * BTILE);
            cur ^= 1;
        }
    }

    // ---- epilogue: lane l16 is N, VGPR v + 8*half is M ----
#pragma unroll
    for (int s = 0; s < 2; ++s) {
        const int mbase = m0 + wm + s * 16 + half * 8;
        float bv[8];
        if (BIAS) {        // 8 consecutive, 32B-aligned floats -> 2 x b128
            float4 b0 = *(const float4*)&bias[mbase];
            float4 b1 = *(const float4*)&bias[mbase + 4];
            bv[0] = b0.x; bv[1] = b0.y; bv[2] = b0.z; bv[3] = b0.w;
            bv[4] = b1.x; bv[5] = b1.y; bv[6] = b1.z; bv[7] = b1.w;
        }
#pragma unroll
        for (int t = 0; t < 4; ++t) {
            const int n = n0 + wn + t * 16 + l16;
#pragma unroll
            for (int v = 0; v < 8; ++v) {
                const int m = mbase + v;
                float val = acc[s][t][v];
                if (BIAS) val += bv[v];
                if (OUT_BF16) {
                    unsigned short* D = (unsigned short*)Dg;
                    D[(size_t)batch * strideD + (size_t)m * ldd + n] = f2bf(val);
                } else {
                    float* D = (float*)Dg;
                    D[(size_t)batch * strideD + (size_t)m * ldd + n] = val;
                }
            }
        }
    }
}

// ---------------------------------------------------------------------------
// Row softmax: f (f32, [rows, 512]) -> attn (bf16, [rows, 512])
// ---------------------------------------------------------------------------
__global__ __launch_bounds__(256) void softmax_kernel(
    const float* __restrict__ f, unsigned short* __restrict__ attn)
{
    const long long row = blockIdx.x;
    const float* fr = f + row * NN;
    unsigned short* ar = attn + row * NN;

    __shared__ float red[256];
    const int tid = threadIdx.x;

    float v0 = fr[tid];
    float v1 = fr[tid + 256];

    red[tid] = fmaxf(v0, v1);
    __syncthreads();
#pragma unroll
    for (int s = 128; s > 0; s >>= 1) {
        if (tid < s) red[tid] = fmaxf(red[tid], red[tid + s]);
        __syncthreads();
    }
    const float rmax = red[0];
    __syncthreads();

    const float e0 = __expf(v0 - rmax);
    const float e1 = __expf(v1 - rmax);
    red[tid] = e0 + e1;
    __syncthreads();
#pragma unroll
    for (int s = 128; s > 0; s >>= 1) {
        if (tid < s) red[tid] += red[tid + s];
        __syncthreads();
    }
    const float inv = 1.0f / red[0];

    ar[tid]       = f2bf(e0 * inv);
    ar[tid + 256] = f2bf(e1 * inv);
}

// ---------------------------------------------------------------------------
// BatchNorm stats: one block per channel c, reduce over B x N = 32768 values
// ---------------------------------------------------------------------------
__global__ __launch_bounds__(256) void bn_stats_kernel(
    const unsigned short* __restrict__ wy,
    float* __restrict__ mean, float* __restrict__ rstd)
{
    const int c   = blockIdx.x;
    const int tid = threadIdx.x;

    float s = 0.f, ss = 0.f;
    for (int b = 0; b < BB; ++b) {
        const unsigned short* p = wy + ((size_t)b * CC + c) * NN;
        for (int n = tid; n < NN; n += 256) {
            float v = bf2f(p[n]);
            s  += v;
            ss += v * v;
        }
    }

    __shared__ float rs[256];
    __shared__ float rq[256];
    rs[tid] = s; rq[tid] = ss;
    __syncthreads();
#pragma unroll
    for (int k = 128; k > 0; k >>= 1) {
        if (tid < k) { rs[tid] += rs[tid + k]; rq[tid] += rq[tid + k]; }
        __syncthreads();
    }
    if (tid == 0) {
        const float cnt = (float)(BB * NN);
        const float mu  = rs[0] / cnt;
        const float var = rq[0] / cnt - mu * mu;
        mean[c] = mu;
        rstd[c] = rsqrtf(var + BN_EPS);
    }
}

// ---------------------------------------------------------------------------
// BN apply + residual: out[b,c,n] = (wy - mean[c])*rstd[c]*gamma[c] + beta[c] + x
// ---------------------------------------------------------------------------
__global__ __launch_bounds__(256) void bn_apply_kernel(
    const unsigned short* __restrict__ wy, const float* __restrict__ x,
    const float* __restrict__ mean, const float* __restrict__ rstd,
    const float* __restrict__ gamma, const float* __restrict__ beta,
    float* __restrict__ out)
{
    const long long i = (long long)blockIdx.x * 256 + threadIdx.x;
    const long long total = (long long)BB * CC * NN;
    if (i < total) {
        const int c = (int)((i / NN) % CC);
        out[i] = (bf2f(wy[i]) - mean[c]) * rstd[c] * gamma[c] + beta[c] + x[i];
    }
}

// ---------------------------------------------------------------------------
// Host-side orchestration
// ---------------------------------------------------------------------------
extern "C" void kernel_launch(void* const* d_in, const int* in_sizes, int n_in,
                              void* d_out, int out_size, void* d_ws, size_t ws_size,
                              hipStream_t stream) {
    (void)in_sizes; (void)n_in; (void)out_size; (void)ws_size;

    const float* x     = (const float*)d_in[0];
    const float* Wg    = (const float*)d_in[1];
    const float* bg    = (const float*)d_in[2];
    const float* Wt    = (const float*)d_in[3];
    const float* bt    = (const float*)d_in[4];
    const float* Wp    = (const float*)d_in[5];
    const float* bp    = (const float*)d_in[6];
    const float* Wo    = (const float*)d_in[7];
    const float* bo    = (const float*)d_in[8];
    const float* gamma = (const float*)d_in[9];
    const float* beta  = (const float*)d_in[10];
    float* out = (float*)d_out;

    // ---- workspace layout (256-B aligned slices) ----
    char* ws = (char*)d_ws;
    size_t off = 0;
    auto take = [&](size_t bytes) -> char* {
        char* p = ws + off;
        off += (bytes + 255) & ~(size_t)255;
        return p;
    };
    const size_t xN  = (size_t)BB * CC * NN;   // 67.1M
    const size_t pN  = (size_t)BB * CI * NN;   // 33.6M
    const size_t aN  = (size_t)BB * NN * NN;   // 16.8M
    unsigned short* xb   = (unsigned short*)take(xN * 2);
    unsigned short* wgb  = (unsigned short*)take((size_t)CI * CC * 2);
    unsigned short* wtb  = (unsigned short*)take((size_t)CI * CC * 2);
    unsigned short* wpb  = (unsigned short*)take((size_t)CI * CC * 2);
    unsigned short* wob  = (unsigned short*)take((size_t)CC * CI * 2);
    unsigned short* gb   = (unsigned short*)take(pN * 2);
    unsigned short* thb  = (unsigned short*)take(pN * 2);
    unsigned short* phb  = (unsigned short*)take(pN * 2);
    float*          ff   = (float*)take(aN * 4);
    unsigned short* attn = (unsigned short*)take(aN * 2);
    unsigned short* yb   = (unsigned short*)take(pN * 2);
    unsigned short* wyb  = (unsigned short*)take(xN * 2);
    float*          mean = (float*)take((size_t)CC * 4);
    float*          rstd = (float*)take((size_t)CC * 4);

    // ---- 1) f32 -> bf16 conversions ----
    auto cvt = [&](const float* src, unsigned short* dst, long long n) {
        cvt_f32_bf16_kernel<<<(unsigned)((n + 1023) / 1024), 256, 0, stream>>>(src, dst, n);
    };
    cvt(x,  xb,  (long long)xN);
    cvt(Wg, wgb, (long long)CI * CC);
    cvt(Wt, wtb, (long long)CI * CC);
    cvt(Wp, wpb, (long long)CI * CC);
    cvt(Wo, wob, (long long)CC * CI);

    const long long sX = (long long)CC * NN;  // x batch stride
    const long long sP = (long long)CI * NN;  // projection batch stride
    const long long sA = (long long)NN * NN;  // attention batch stride

    // ---- 2) projections: {g,th,ph}[b] = W * x[b] + bias   (M=1024,N=512,K=2048)
    {
        dim3 grid(NN / TNB, CI / TMB, BB);
        gemm_wmma_kernel<0, 0, 1, 1><<<grid, 256, 0, stream>>>(
            wgb, 0, CC, xb, sX, NN, gb, sP, NN, bg, CI, NN, CC);
        gemm_wmma_kernel<0, 0, 1, 1><<<grid, 256, 0, stream>>>(
            wtb, 0, CC, xb, sX, NN, thb, sP, NN, bt, CI, NN, CC);
        gemm_wmma_kernel<0, 0, 1, 1><<<grid, 256, 0, stream>>>(
            wpb, 0, CC, xb, sX, NN, phb, sP, NN, bp, CI, NN, CC);
    }

    // ---- 3) f[b] = th[b]^T * ph[b]   (M=512,N=512,K=1024, A transposed, f32 out)
    {
        dim3 grid(NN / TNB, NN / TMB, BB);
        gemm_wmma_kernel<1, 0, 0, 0><<<grid, 256, 0, stream>>>(
            thb, sP, NN, phb, sP, NN, ff, sA, NN, nullptr, NN, NN, CI);
    }

    // ---- 4) softmax over last axis, f32 -> bf16 ----
    softmax_kernel<<<(unsigned)(BB * NN), 256, 0, stream>>>(ff, attn);

    // ---- 5) y[b] = g[b] * attn[b]^T  (M=1024,N=512,K=512, B transposed)
    {
        dim3 grid(NN / TNB, CI / TMB, BB);
        gemm_wmma_kernel<0, 1, 1, 0><<<grid, 256, 0, stream>>>(
            gb, sP, NN, attn, sA, NN, yb, sP, NN, nullptr, CI, NN, NN);
    }

    // ---- 6) wy[b] = Wo * y[b] + bo  (M=2048,N=512,K=1024, bf16 out)
    {
        dim3 grid(NN / TNB, CC / TMB, BB);
        gemm_wmma_kernel<0, 0, 1, 1><<<grid, 256, 0, stream>>>(
            wob, 0, CI, yb, sP, NN, wyb, sX, NN, bo, CC, NN, CI);
    }

    // ---- 7) BatchNorm stats over (B, N) per channel ----
    bn_stats_kernel<<<CC, 256, 0, stream>>>(wyb, mean, rstd);

    // ---- 8) BN apply + residual -> f32 output ----
    {
        const long long total = (long long)BB * CC * NN;
        bn_apply_kernel<<<(unsigned)((total + 255) / 256), 256, 0, stream>>>(
            wyb, x, mean, rstd, gamma, beta, out);
    }
}
